// StructureAwareAttention_50757923504212
// MI455X (gfx1250) — compile-verified, matrix operand order checked by reading, not run on hardware
//
#include <hip/hip_runtime.h>

typedef __bf16 bf16;
typedef __attribute__((ext_vector_type(16))) __bf16 v16bf;
typedef __attribute__((ext_vector_type(8)))  __bf16 v8bf;
typedef __attribute__((ext_vector_type(8)))  float  v8f;
typedef __attribute__((ext_vector_type(4)))  float  v4f;

#define WMMA_BF16(a,b,c) \
  __builtin_amdgcn_wmma_f32_16x16x32_bf16(false,(a),false,(b),(short)0,(c),false,false)

// LDS layout (bytes). Row strides chosen as multiples of 32B for aligned v16bf loads.
constexpr size_t OFF_Q   = 0;                       // bf16 [256][112]
constexpr size_t OFF_K   = OFF_Q  + 256*112*2;      // bf16 [256][112]
constexpr size_t OFF_VT  = OFF_K  + 256*112*2;      // bf16 [96][272]   (V transposed)
constexpr size_t OFF_SCR = OFF_VT + 96*272*2;       // union{ bf16 xT[96][272]; f32 S[128][272] }
constexpr size_t OFF_VEC = OFF_SCR + 128*272*4;     // f32 vectors (7*256) + yup row (96, pad 128)
constexpr size_t SMEM_BYTES = OFF_VEC + 1920*4;     // = 313856 <= 320KB
static_assert(SMEM_BYTES <= 320u*1024u, "LDS budget");

// ---------------------------------------------------------------------------
// Prep: cast weights f32->bf16; fold the y branch into per-channel vectors:
//   aq = wq@wy,  cq = wq@by + bq   (same for k)
// ---------------------------------------------------------------------------
__global__ __launch_bounds__(256)
void prep_kernel(const float* __restrict__ wq, const float* __restrict__ wk,
                 const float* __restrict__ wv, const float* __restrict__ wy,
                 const float* __restrict__ by, const float* __restrict__ bq,
                 const float* __restrict__ bk,
                 bf16* __restrict__ wqb, bf16* __restrict__ wkb, bf16* __restrict__ wvb,
                 float* __restrict__ aq, float* __restrict__ ak,
                 float* __restrict__ cq, float* __restrict__ ck) {
  int i = blockIdx.x * 256 + threadIdx.x;   // grid = 256 blocks -> 65536 elements
  wqb[i] = (bf16)wq[i];
  wkb[i] = (bf16)wk[i];
  wvb[i] = (bf16)wv[i];
  if (blockIdx.x == 0) {
    int o = threadIdx.x;
    float sq = 0.f, sk = 0.f, tq = 0.f, tk = 0.f;
    for (int c = 0; c < 256; ++c) {
      float wyc = wy[c], byc = by[c];
      float q = wq[o*256 + c], k = wk[o*256 + c];
      sq += q * wyc;  tq += q * byc;
      sk += k * wyc;  tk += k * byc;
    }
    aq[o] = sq; ak[o] = sk;
    cq[o] = tq + bq[o]; ck[o] = tk + bk[o];
  }
}

// ---------------------------------------------------------------------------
// Fused attention: one workgroup (256 thr, 8 waves) per (b,h) spatial row.
// ---------------------------------------------------------------------------
__global__ __launch_bounds__(256)
void attn_kernel(const float* __restrict__ x, const float* __restrict__ y,
                 const bf16* __restrict__ wqb, const bf16* __restrict__ wkb,
                 const bf16* __restrict__ wvb,
                 const float* __restrict__ aqg, const float* __restrict__ akg,
                 const float* __restrict__ cqg, const float* __restrict__ ckg,
                 const float* __restrict__ bqg, const float* __restrict__ bkg,
                 const float* __restrict__ bvg,
                 float* __restrict__ out) {
  extern __shared__ v16bf smem_raw[];           // 32B-aligned dynamic LDS
  char* smem = (char*)smem_raw;
  bf16*  sQ   = (bf16*)(smem + OFF_Q);          // [c][112] row-major (w)
  bf16*  sK   = (bf16*)(smem + OFF_K);          // [d][112]
  bf16*  sVT  = (bf16*)(smem + OFF_VT);         // [w][272] (= V transposed)
  float* sS   = (float*)(smem + OFF_SCR);       // scores [128][272] f32 (per pass)
  bf16*  sXT  = (bf16*)(smem + OFF_SCR);        // xT [w][272] bf16 (aliased, dead after GEMM1)
  float* sVec = (float*)(smem + OFF_VEC);
  float* sAq = sVec;          float* sAk = sVec + 256;
  float* sCq = sVec + 512;    float* sCk = sVec + 768;
  float* sBq = sVec + 1024;   float* sBk = sVec + 1280;
  float* sBv = sVec + 1536;   float* sYup = sVec + 1792;

  const int tid = threadIdx.x;
  const int bh  = blockIdx.x;
  const int b = bh / 96, h = bh % 96;

  // --- per-channel vectors ---
  sAq[tid] = aqg[tid]; sAk[tid] = akg[tid];
  sCq[tid] = cqg[tid]; sCk[tid] = ckg[tid];
  sBq[tid] = bqg[tid]; sBk[tid] = bkg[tid]; sBv[tid] = bvg[tid];

  // --- bilinear upsample (half-pixel) of the 1-channel y for this output row ---
  if (tid < 96) {
    float si = (h + 0.5f) * 0.25f - 0.5f;
    int   i0 = (int)floorf(si);
    float fi = si - (float)i0;
    int i0c = min(max(i0, 0), 23), i1c = min(max(i0 + 1, 0), 23);
    float sj = (tid + 0.5f) * 0.25f - 0.5f;
    int   j0 = (int)floorf(sj);
    float fj = sj - (float)j0;
    int j0c = min(max(j0, 0), 23), j1c = min(max(j0 + 1, 0), 23);
    const float* yb = y + b * 576;
    float v00 = yb[i0c*24 + j0c], v01 = yb[i0c*24 + j1c];
    float v10 = yb[i1c*24 + j0c], v11 = yb[i1c*24 + j1c];
    sYup[tid] = (v00*(1.f-fj) + v01*fj)*(1.f-fi) + (v10*(1.f-fj) + v11*fj)*fi;
  }

  // --- load x[b,:,h,:] and store transposed bf16 xT[w][c] (thread = channel) ---
  {
    const float* xr = x + ((size_t)(b*256 + tid) * 96 + h) * 96;
    #pragma unroll
    for (int w0 = 0; w0 < 96; w0 += 4) {
      v4f xv = *(const v4f*)(xr + w0);
      sXT[(w0+0)*272 + tid] = (bf16)xv.x;
      sXT[(w0+1)*272 + tid] = (bf16)xv.y;
      sXT[(w0+2)*272 + tid] = (bf16)xv.z;
      sXT[(w0+3)*272 + tid] = (bf16)xv.w;
    }
  }
  __syncthreads();

  const int wave = tid >> 5, lane = tid & 31;
  const int lr16 = lane & 15, lh = lane >> 4;   // WMMA lane decomposition

  // --- GEMM1: {Q,K,V} projections. 48 (matrix,mtile) jobs over 8 waves ---
  for (int job = wave; job < 48; job += 8) {
    const int mat = job >> 4, mt = job & 15;
    const bf16* Wm = (mat == 0) ? wqb : ((mat == 1) ? wkb : wvb);
    v8f acc[6] = {};
    #pragma unroll
    for (int kt = 0; kt < 8; ++kt) {
      v16bf a = *(const v16bf*)(Wm + (mt*16 + lr16)*256 + kt*32 + lh*16);
      #pragma unroll
      for (int nt = 0; nt < 6; ++nt) {
        v16bf bb = *(const v16bf*)(sXT + (nt*16 + lr16)*272 + kt*32 + lh*16);
        acc[nt] = WMMA_BF16(a, bb, acc[nt]);
      }
    }
    if (mat < 2) {  // Q/K: gate with rank-1 y branch, store row-major [c][w]
      const float* av = mat == 0 ? sAq : sAk;
      const float* cv = mat == 0 ? sCq : sCk;
      const float* bbv = mat == 0 ? sBq : sBk;
      bf16* dst = mat == 0 ? sQ : sK;
      #pragma unroll
      for (int nt = 0; nt < 6; ++nt) {
        int n = nt*16 + lr16;
        float yv = sYup[n];
        #pragma unroll
        for (int r = 0; r < 8; ++r) {
          int m = mt*16 + lh*8 + r;
          float xp = acc[nt][r] + bbv[m];
          float yp = av[m] * yv + cv[m];
          dst[m*112 + n] = (bf16)(xp * yp);
        }
      }
    } else {        // V: store transposed [w][c] (contiguous 16B per lane)
      #pragma unroll
      for (int nt = 0; nt < 6; ++nt) {
        int n = nt*16 + lr16;
        int mbase = mt*16 + lh*8;
        v8bf tv;
        #pragma unroll
        for (int r = 0; r < 8; ++r) tv[r] = (bf16)(acc[nt][r] + sBv[mbase + r]);
        *(v8bf*)(sVT + n*272 + mbase) = tv;
      }
    }
  }
  __syncthreads();

  // --- attention: two 128-row passes; each wave owns 16 score rows ---
  const float scale = 0.17677669529663689f;  // 1/sqrt(32)
  for (int rb = 0; rb < 2; ++rb) {
    const int mt = rb*8 + wave;              // score row tile (rows mt*16..+15)

    // GEMM2: S = Q @ K^T  (K dim = w = 96)
    v8f sacc[16] = {};
    #pragma unroll
    for (int kt = 0; kt < 3; ++kt) {
      v16bf a = *(const v16bf*)(sQ + (mt*16 + lr16)*112 + kt*32 + lh*16);
      #pragma unroll
      for (int nt = 0; nt < 16; ++nt) {
        v16bf bb = *(const v16bf*)(sK + (nt*16 + lr16)*112 + kt*32 + lh*16);
        sacc[nt] = WMMA_BF16(a, bb, sacc[nt]);
      }
    }
    float* srow = sS + wave*16*272;
    #pragma unroll
    for (int nt = 0; nt < 16; ++nt)
      #pragma unroll
      for (int r = 0; r < 8; ++r)
        srow[(lh*8 + r)*272 + nt*16 + lr16] = sacc[nt][r] * scale;
    __syncthreads();

    // softmax over 256 cols; overwrite rows in place with bf16 probs
    for (int lr = 0; lr < 16; ++lr) {
      float* row = sS + (wave*16 + lr)*272;
      float v[8]; float mx = -3.0e38f;
      #pragma unroll
      for (int j = 0; j < 8; ++j) { v[j] = row[lane*8 + j]; mx = fmaxf(mx, v[j]); }
      #pragma unroll
      for (int off = 16; off > 0; off >>= 1) mx = fmaxf(mx, __shfl_xor(mx, off, 32));
      float s = 0.f;
      #pragma unroll
      for (int j = 0; j < 8; ++j) { v[j] = __expf(v[j] - mx); s += v[j]; }
      #pragma unroll
      for (int off = 16; off > 0; off >>= 1) s += __shfl_xor(s, off, 32);
      float inv = 1.f / s;
      v8bf pv;
      #pragma unroll
      for (int j = 0; j < 8; ++j) pv[j] = (bf16)(v[j] * inv);
      *(v8bf*)((bf16*)row + lane*8) = pv;
    }
    __syncthreads();

    // GEMM3: out_rows = P @ V   (K dim = d = 256)
    v8f oacc[6] = {};
    const bf16* Pb = (const bf16*)(sS + wave*16*272);  // row stride 544 bf16
    #pragma unroll
    for (int kt = 0; kt < 8; ++kt) {
      v16bf a = *(const v16bf*)(Pb + lr16*544 + kt*32 + lh*16);
      #pragma unroll
      for (int nt = 0; nt < 6; ++nt) {
        v16bf bb = *(const v16bf*)(sVT + (nt*16 + lr16)*272 + kt*32 + lh*16);
        oacc[nt] = WMMA_BF16(a, bb, oacc[nt]);
      }
    }
    #pragma unroll
    for (int nt = 0; nt < 6; ++nt) {
      int n = nt*16 + lr16;
      #pragma unroll
      for (int r = 0; r < 8; ++r) {
        int m = mt*16 + lh*8 + r;
        out[((size_t)b*256 + m)*9216 + (size_t)h*96 + n] = oacc[nt][r];
      }
    }
    __syncthreads();
  }
}

extern "C" void kernel_launch(void* const* d_in, const int* in_sizes, int n_in,
                              void* d_out, int out_size, void* d_ws, size_t ws_size,
                              hipStream_t stream) {
  const float* x  = (const float*)d_in[0];
  const float* y  = (const float*)d_in[1];
  const float* wy = (const float*)d_in[2];
  const float* by = (const float*)d_in[3];
  const float* wq = (const float*)d_in[4];
  const float* bq = (const float*)d_in[5];
  const float* wk = (const float*)d_in[6];
  const float* bk = (const float*)d_in[7];
  const float* wv = (const float*)d_in[8];
  const float* bv = (const float*)d_in[9];
  float* out = (float*)d_out;

  // workspace carve: bf16 weights (3x64K) + 4 folded vectors
  bf16* wqb = (bf16*)d_ws;
  bf16* wkb = wqb + 65536;
  bf16* wvb = wkb + 65536;
  float* aq = (float*)(wvb + 65536);
  float* ak = aq + 256;
  float* cq = ak + 256;
  float* ck = cq + 256;

  prep_kernel<<<256, 256, 0, stream>>>(wq, wk, wv, wy, by, bq, bk,
                                       wqb, wkb, wvb, aq, ak, cq, ck);

  hipFuncSetAttribute(reinterpret_cast<const void*>(attn_kernel),
                      hipFuncAttributeMaxDynamicSharedMemorySize,
                      (int)SMEM_BYTES);
  attn_kernel<<<768, 256, SMEM_BYTES, stream>>>(x, y, wqb, wkb, wvb,
                                                aq, ak, cq, ck, bq, bk, bv, out);
  (void)in_sizes; (void)n_in; (void)out_size; (void)ws_size;
}
